// ViT_80255758893115
// MI455X (gfx1250) — matrix operand order repaired and checked by reading, not compiled, not run on hardware
//
#include <hip/hip_runtime.h>

// ---------------- problem constants ----------------
#define BATCH 64
#define HWPIX 50176          // 224*224
#define SSEG  196
#define SEQ   197
#define NDIM  768
#define NQKV  2304
#define NMLP  3072
#define NHEADS 12
#define DHEAD 64
#define NCLSR 1000
#define NCLSP 1024
#define MROWS (BATCH*SEQ)    // 12608 = 197*64
#define DEPTH 2

// ---------------- WMMA types (CDNA5, wave32) ----------------
typedef __attribute__((ext_vector_type(16))) __bf16 v16bf;
typedef __attribute__((ext_vector_type(8)))  float  v8f;

__device__ __forceinline__ unsigned short f2bf(float f) {
  unsigned int u = __float_as_uint(f);
  unsigned int r = u + 0x7FFFu + ((u >> 16) & 1u);   // round-to-nearest-even
  return (unsigned short)(r >> 16);
}
__device__ __forceinline__ float bf2f(unsigned short v) {
  return __uint_as_float(((unsigned int)v) << 16);
}
__device__ __forceinline__ float gelu_tanh(float x) {
  return 0.5f * x * (1.0f + tanhf(0.7978845608028654f * (x + 0.044715f * x * x * x)));
}
// dot of one packed bf16 pair with two fp32 values
__device__ __forceinline__ float dp2(unsigned p, float q0, float q1) {
  return q0 * bf2f((unsigned short)(p & 0xffffu)) + q1 * bf2f((unsigned short)(p >> 16));
}

// CDNA5 async global->LDS copy (ASYNCcnt-tracked); vdst = 32-bit LDS byte addr
__device__ __forceinline__ void async_load_b128(unsigned lds_addr, const void* gptr) {
  asm volatile("global_load_async_to_lds_b128 %0, %1, off"
               :: "v"(lds_addr), "v"((unsigned long long)(size_t)gptr)
               : "memory");
}
__device__ __forceinline__ void wait_async_le2() {
  asm volatile("s_wait_asynccnt 0x2" ::: "memory");
}
__device__ __forceinline__ void wait_async_le1() {
  asm volatile("s_wait_asynccnt 0x1" ::: "memory");
}
__device__ __forceinline__ void wait_async_le0() {
  asm volatile("s_wait_asynccnt 0x0" ::: "memory");
}

// ---------------- weight pack: fp32 (K x Ns) -> bf16 WMMA-fragment tiles ------
// Tile (kt,nt): K rows [kt*32,+32) x cols [nt*16,+16) stored as 512 elements;
// lane L owns 16 contiguous bf16, element e = W[kt*32+(L>>4)*8+(e&7)+(e>>3)*16][nt*16+(L&15)]
__global__ void pack_w_kernel(const float* __restrict__ src,
                              unsigned short* __restrict__ dst,
                              int K, int Ns, int Nd) {
  long long i = (long long)blockIdx.x * 256 + threadIdx.x;
  long long tot = (long long)(Nd / 16) * (K / 32) * 512;
  if (i >= tot) return;
  int r = (int)(i & 511);
  long long t = i >> 9;
  int kt = (int)(t % (K / 32));
  int nt = (int)(t / (K / 32));
  int lane = r >> 4, e = r & 15;
  int col = nt * 16 + (lane & 15);
  int k   = kt * 32 + (lane >> 4) * 8 + (e & 7) + (e >> 3) * 16;
  dst[i] = (col < Ns) ? f2bf(src[(long long)k * Ns + col]) : (unsigned short)0;
}

// ---------------- fused segment reduce of raw image (B,C,H,W) -----------------
__global__ __launch_bounds__(256) void seg_reduce_kernel(
    const float* __restrict__ img, const int* __restrict__ lab,
    float* __restrict__ seg) {
  __shared__ float s[SSEG * 4];
  const int b = blockIdx.x, tid = threadIdx.x;
  for (int i = tid; i < SSEG * 4; i += 256) s[i] = 0.f;
  __syncthreads();
  const int* lb = lab + (long long)b * HWPIX;
  const float* i0 = img + ((long long)b * 3 + 0) * HWPIX;
  const float* i1 = img + ((long long)b * 3 + 1) * HWPIX;
  const float* i2 = img + ((long long)b * 3 + 2) * HWPIX;
  for (int p = tid; p < HWPIX; p += 256) {
    int l = lb[p];
    atomicAdd(&s[l * 4 + 0], i0[p]);
    atomicAdd(&s[l * 4 + 1], i1[p]);
    atomicAdd(&s[l * 4 + 2], i2[p]);
    atomicAdd(&s[l * 4 + 3], 1.0f);
  }
  __syncthreads();
  for (int i = tid; i < SSEG * 4; i += 256)
    seg[(long long)b * SSEG * 4 + i] = s[i];
}

// ---------------- x0 = concat(cls, segmean @ W_feat) + pos  (fp32 residual) ---
__global__ void build_x0_kernel(const float* __restrict__ seg,
                                const float* __restrict__ Wf,
                                const float* __restrict__ pos,
                                const float* __restrict__ cls,
                                float* __restrict__ x) {
  long long i = (long long)blockIdx.x * 256 + threadIdx.x;
  if (i >= (long long)MROWS * NDIM) return;
  int d = (int)(i % NDIM);
  int r = (int)(i / NDIM);
  int s = r % SEQ, b = r / SEQ;
  float val;
  if (s == 0) {
    val = cls[d];
  } else {
    const float* sg = seg + ((long long)b * SSEG + (s - 1)) * 4;
    float cnt = fmaxf(sg[3], 1.0f);
    val = (sg[0] * Wf[d] + sg[1] * Wf[NDIM + d] + sg[2] * Wf[2 * NDIM + d]) / cnt;
  }
  x[i] = val + pos[(long long)s * NDIM + d];
}

// ---------------- LayerNorm (fp32 in, bf16 out), one row per block ----------
// single pass: sum + sum-of-squares, wave32 shuffle reduce, one barrier
__global__ __launch_bounds__(256) void ln_kernel(const float* __restrict__ x,
                                                 long long rowStride,
                                                 const float* __restrict__ g,
                                                 const float* __restrict__ bb,
                                                 unsigned short* __restrict__ out) {
  __shared__ float ws[8], ws2[8];
  const int tid = threadIdx.x, lane = tid & 31, wave = tid >> 5;
  const float* xr = x + (long long)blockIdx.x * rowStride;
  float v0 = xr[tid], v1 = xr[tid + 256], v2 = xr[tid + 512];
  float s  = v0 + v1 + v2;
  float s2 = v0 * v0 + v1 * v1 + v2 * v2;
#pragma unroll
  for (int m = 16; m > 0; m >>= 1) {
    s  += __shfl_xor(s,  m, 32);
    s2 += __shfl_xor(s2, m, 32);
  }
  if (lane == 0) { ws[wave] = s; ws2[wave] = s2; }
  __syncthreads();
  float st  = ws[0] + ws[1] + ws[2] + ws[3] + ws[4] + ws[5] + ws[6] + ws[7];
  float s2t = ws2[0] + ws2[1] + ws2[2] + ws2[3] + ws2[4] + ws2[5] + ws2[6] + ws2[7];
  float mean = st * (1.0f / NDIM);
  float rstd = rsqrtf(s2t * (1.0f / NDIM) - mean * mean + 1e-5f);
  unsigned short* orow = out + (long long)blockIdx.x * NDIM;
  orow[tid]       = f2bf((v0 - mean) * rstd * g[tid]       + bb[tid]);
  orow[tid + 256] = f2bf((v1 - mean) * rstd * g[tid + 256] + bb[tid + 256]);
  orow[tid + 512] = f2bf((v2 - mean) * rstd * g[tid + 512] + bb[tid + 512]);
}

// ---------------- bf16 WMMA GEMM: C = act(A@Wpk + bias) (+resid) -------------
// 256 threads = 8 waves; wave = 16-col strip x 64 rows (4 accumulators).
// A tile (64x32) triple-buffered in LDS via async global->LDS DMA, two tiles
// in flight. B = contiguous packed-fragment b128 stream, one step ahead.
#define ALDS_STRIDE 40             // 80B row stride: 16B-aligned, conflict-free
#define ABUF (64 * ALDS_STRIDE)    // elements per buffer
#define ABUF_BYTES (ABUF * 2)
union AFrag { uint4 q[2]; v16bf v; };

__global__ __launch_bounds__(256) void gemm_bf16_kernel(
    const unsigned short* __restrict__ A, const unsigned short* __restrict__ Bpk,
    const float* __restrict__ bias, const float* __restrict__ resid,
    float* __restrict__ Cf, unsigned short* __restrict__ Cb,
    int M, int N, int K, int lda, int ldc, int n_real, int act) {
  __shared__ unsigned short Alds[3 * ABUF];
  const int tid  = threadIdx.x;
  const int lane = tid & 31;
  const int wave = tid >> 5;
  const int half = lane >> 4;
  const int l15  = lane & 15;
  const int m_base = blockIdx.y * 64;
  const int n0 = blockIdx.x * 128 + wave * 16;
  (void)M; (void)N;

  v8f acc0 = {0,0,0,0,0,0,0,0}, acc1 = acc0, acc2 = acc0, acc3 = acc0;

  // cooperative A-stage: thread t copies row t>>2, 8-elem segment t&3
  const int arow = tid >> 2, aseg = tid & 3;
  const unsigned short* agp = A + (long long)(m_base + arow) * lda + aseg * 8;
  const unsigned lbase =
      (unsigned)(size_t)(void*)&Alds[arow * ALDS_STRIDE + aseg * 8];
  const unsigned short* bbase =
      Bpk + ((long long)(n0 >> 4) * (K >> 5)) * 512 + lane * 16;
  const int KT = K >> 5;   // >= 24 for all our shapes

  // prologue: two tiles in flight, first B fragment
  async_load_b128(lbase, agp);
  async_load_b128(lbase + ABUF_BYTES, agp + 32);
  AFrag bcur;
  bcur.q[0] = *(const uint4*)(bbase);
  bcur.q[1] = *(const uint4*)(bbase + 8);

  auto body = [&](int kt, int bufIdx) {
    AFrag bnxt = bcur;
    if (kt + 1 < KT) {
      const unsigned short* bp = bbase + (long long)(kt + 1) * 512;
      bnxt.q[0] = *(const uint4*)(bp);
      bnxt.q[1] = *(const uint4*)(bp + 8);
      __builtin_prefetch((const void*)(bp + 512), 0, 3);  // speculative ok
    }
    const unsigned short* abuf = &Alds[bufIdx * ABUF];
    AFrag af0, af1, af2, af3;
    const unsigned short* ap0 = abuf + l15 * ALDS_STRIDE + half * 8;
    const unsigned short* ap1 = ap0 + 16 * ALDS_STRIDE;
    const unsigned short* ap2 = ap0 + 32 * ALDS_STRIDE;
    const unsigned short* ap3 = ap0 + 48 * ALDS_STRIDE;
    af0.q[0] = *(const uint4*)(ap0); af0.q[1] = *(const uint4*)(ap0 + 16);
    af1.q[0] = *(const uint4*)(ap1); af1.q[1] = *(const uint4*)(ap1 + 16);
    af2.q[0] = *(const uint4*)(ap2); af2.q[1] = *(const uint4*)(ap2 + 16);
    af3.q[0] = *(const uint4*)(ap3); af3.q[1] = *(const uint4*)(ap3 + 16);
    acc0 = __builtin_amdgcn_wmma_f32_16x16x32_bf16(false, af0.v, false, bcur.v, (short)0, acc0, false, false);
    acc1 = __builtin_amdgcn_wmma_f32_16x16x32_bf16(false, af1.v, false, bcur.v, (short)0, acc1, false, false);
    acc2 = __builtin_amdgcn_wmma_f32_16x16x32_bf16(false, af2.v, false, bcur.v, (short)0, acc2, false, false);
    acc3 = __builtin_amdgcn_wmma_f32_16x16x32_bf16(false, af3.v, false, bcur.v, (short)0, acc3, false, false);
    bcur = bnxt;
  };

  // steady state: branch-free async issue + wait<=2 (tile kt retired)
  int get = 0;
  int kt = 0;
  for (; kt + 2 < KT; ++kt) {
    __syncthreads();                 // all waves done reading buffer being reused
    int put = get + 2; if (put >= 3) put -= 3;
    async_load_b128(lbase + (unsigned)(put * ABUF_BYTES),
                    agp + (long long)(kt + 2) * 32);
    wait_async_le2();
    __syncthreads();                 // tile kt visible to all waves
    body(kt, get);
    if (++get == 3) get = 0;
  }
  // epilogue: tiles KT-2, KT-1 (no further issues)
  wait_async_le1();
  __syncthreads();
  body(kt, get);
  ++kt; if (++get == 3) get = 0;
  wait_async_le0();
  __syncthreads();
  body(kt, get);

  const int n = n0 + l15;
  if (n < n_real) {
#define STORE_MT(MT, ACC)                                           \
    _Pragma("unroll")                                               \
    for (int v = 0; v < 8; ++v) {                                   \
      int m = m_base + (MT) * 16 + half * 8 + v;                    \
      float r = ACC[v];                                             \
      if (bias)     r += bias[n];                                   \
      if (act == 1) r = gelu_tanh(r);                               \
      long long off = (long long)m * ldc + n;                       \
      if (resid)    r += resid[off];                                \
      if (Cb) Cb[off] = f2bf(r);                                    \
      else    Cf[off] = r;                                          \
    }
    STORE_MT(0, acc0)
    STORE_MT(1, acc1)
    STORE_MT(2, acc2)
    STORE_MT(3, acc3)
#undef STORE_MT
  }
}

// ---------------- attention: one workgroup per (batch, head) -----------------
__global__ __launch_bounds__(256) void attn_kernel(const float* __restrict__ qkv,
                                                   unsigned short* __restrict__ o) {
  const int b = blockIdx.x / NHEADS;
  const int h = blockIdx.x % NHEADS;
  __shared__ unsigned short Kt[SEQ * DHEAD];
  __shared__ unsigned short Vt[SEQ * DHEAD];
  __shared__ float qs[8][DHEAD];
  __shared__ float sc[8][SEQ];
  const int tid = threadIdx.x;
  const int lane = tid & 31;
  const int wave = tid >> 5;

  for (int idx = tid; idx < SEQ * DHEAD; idx += 256) {
    int s = idx / DHEAD, d = idx % DHEAD;
    long long row = (long long)(b * SEQ + s) * NQKV;
    Kt[idx] = f2bf(qkv[row + NDIM     + h * DHEAD + d]);
    Vt[idx] = f2bf(qkv[row + 2 * NDIM + h * DHEAD + d]);
  }
  __syncthreads();

  for (int s0 = 0; s0 < 200; s0 += 8) {          // 25 uniform iterations
    for (int idx = tid; idx < 8 * DHEAD; idx += 256) {
      int qi = idx >> 6, d = idx & 63;
      int s = s0 + qi;
      if (s < SEQ)
        qs[qi][d] = qkv[(long long)(b * SEQ + s) * NQKV + h * DHEAD + d];
    }
    __syncthreads();

    const int s = s0 + wave;
    if (s < SEQ) {
      const float* q = qs[wave];
      float mx = -1e30f;
      for (int k = lane; k < SEQ; k += 32) {
        const unsigned* kw = (const unsigned*)(Kt + k * DHEAD);
        float a = 0.f;
#pragma unroll
        for (int c = 0; c < 16; ++c) a += dp2(kw[c], q[2 * c], q[2 * c + 1]);
        a *= 0.125f;                             // DH^-0.5
        sc[wave][k] = a;
        mx = fmaxf(mx, a);
      }
#pragma unroll
      for (int m = 16; m > 0; m >>= 1) mx = fmaxf(mx, __shfl_xor(mx, m, 32));
      float sum = 0.f;
      for (int k = lane; k < SEQ; k += 32) {
        float p = __expf(sc[wave][k] - mx);
        sc[wave][k] = p;
        sum += p;
      }
#pragma unroll
      for (int m = 16; m > 0; m >>= 1) sum += __shfl_xor(sum, m, 32);
      float inv = 1.0f / sum;

      const int d2 = lane * 2;
      float a0 = 0.f, a1 = 0.f;
      for (int k = 0; k < SEQ; ++k) {
        float p = sc[wave][k];
        unsigned pv = *(const unsigned*)(Vt + k * DHEAD + d2);
        a0 += p * bf2f((unsigned short)(pv & 0xffffu));
        a1 += p * bf2f((unsigned short)(pv >> 16));
      }
      unsigned o0 = f2bf(a0 * inv), o1 = f2bf(a1 * inv);
      *(unsigned*)(o + (long long)(b * SEQ + s) * NDIM + h * DHEAD + d2) =
          o0 | (o1 << 16);
    }
    __syncthreads();
  }
}

// ---------------- host side ---------------------------------------------------
extern "C" void kernel_launch(void* const* d_in, const int* in_sizes, int n_in,
                              void* d_out, int out_size, void* d_ws, size_t ws_size,
                              hipStream_t stream) {
  (void)in_sizes; (void)n_in; (void)out_size; (void)ws_size;

  const float* img    = (const float*)d_in[0];
  const int*   labels = (const int*)d_in[1];
  const float* W_feat = (const float*)d_in[2];
  const float* pos    = (const float*)d_in[3];
  const float* cls    = (const float*)d_in[4];
  const float* ln1_g  = (const float*)d_in[5];
  const float* ln1_b  = (const float*)d_in[6];
  const float* qkv_w  = (const float*)d_in[7];
  const float* out_w  = (const float*)d_in[8];
  const float* out_b  = (const float*)d_in[9];
  const float* ln2_g  = (const float*)d_in[10];
  const float* ln2_b  = (const float*)d_in[11];
  const float* fc1_w  = (const float*)d_in[12];
  const float* fc1_b  = (const float*)d_in[13];
  const float* fc2_w  = (const float*)d_in[14];
  const float* fc2_b  = (const float*)d_in[15];
  const float* lnf_g  = (const float*)d_in[16];
  const float* lnf_b  = (const float*)d_in[17];
  const float* head_w = (const float*)d_in[18];
  const float* head_b = (const float*)d_in[19];
  float* outp = (float*)d_out;

  char* w = (char*)d_ws;
  size_t off = 0;
  auto carve = [&](size_t bytes) -> char* {
    off = (off + 255) & ~(size_t)255;
    char* p = w + off;
    off += bytes;
    return p;
  };
  float*          seg   = (float*)carve((size_t)BATCH * SSEG * 4 * sizeof(float));
  float*          x     = (float*)carve((size_t)MROWS * NDIM * sizeof(float));
  unsigned short* hbuf  = (unsigned short*)carve((size_t)MROWS * NDIM * 2);
  float*          qkv   = (float*)carve((size_t)MROWS * NQKV * sizeof(float));
  unsigned short* obuf  = (unsigned short*)carve((size_t)MROWS * NDIM * 2);
  unsigned short* tbuf  = (unsigned short*)carve((size_t)MROWS * NMLP * 2);
  unsigned short* qkvwb = (unsigned short*)carve((size_t)DEPTH * NDIM * NQKV * 2);
  unsigned short* outwb = (unsigned short*)carve((size_t)DEPTH * NDIM * NDIM * 2);
  unsigned short* fc1wb = (unsigned short*)carve((size_t)DEPTH * NDIM * NMLP * 2);
  unsigned short* fc2wb = (unsigned short*)carve((size_t)DEPTH * NMLP * NDIM * 2);
  unsigned short* hdwb  = (unsigned short*)carve((size_t)NDIM * NCLSP * 2);
  unsigned short* acls  = (unsigned short*)carve((size_t)BATCH * NDIM * 2);

  auto pack = [&](const float* s, unsigned short* d, int K, int Ns, int Nd) {
    long long tot = (long long)(Nd / 16) * (K / 32) * 512;
    pack_w_kernel<<<dim3((unsigned)(tot / 256)), dim3(256), 0, stream>>>(s, d, K, Ns, Nd);
  };
  for (int i = 0; i < DEPTH; ++i) {
    pack(qkv_w + (long long)i * NDIM * NQKV, qkvwb + (long long)i * NDIM * NQKV, NDIM, NQKV, NQKV);
    pack(out_w + (long long)i * NDIM * NDIM, outwb + (long long)i * NDIM * NDIM, NDIM, NDIM, NDIM);
    pack(fc1_w + (long long)i * NDIM * NMLP, fc1wb + (long long)i * NDIM * NMLP, NDIM, NMLP, NMLP);
    pack(fc2_w + (long long)i * NMLP * NDIM, fc2wb + (long long)i * NMLP * NDIM, NMLP, NDIM, NDIM);
  }
  pack(head_w, hdwb, NDIM, NCLSR, NCLSP);

  seg_reduce_kernel<<<dim3(BATCH), dim3(256), 0, stream>>>(img, labels, seg);
  build_x0_kernel<<<dim3((unsigned)(((long long)MROWS * NDIM + 255) / 256)), dim3(256), 0, stream>>>(
      seg, W_feat, pos, cls, x);

  auto gemm = [&](const unsigned short* A, const unsigned short* Bpk,
                  const float* bias, const float* resid, float* Cf, unsigned short* Cb,
                  int M, int N, int K, int lda, int ldc, int n_real, int act) {
    gemm_bf16_kernel<<<dim3((unsigned)(N / 128), (unsigned)(M / 64)), dim3(256), 0, stream>>>(
        A, Bpk, bias, resid, Cf, Cb, M, N, K, lda, ldc, n_real, act);
  };

  for (int i = 0; i < DEPTH; ++i) {
    ln_kernel<<<dim3(MROWS), dim3(256), 0, stream>>>(x, (long long)NDIM,
        ln1_g + (long long)i * NDIM, ln1_b + (long long)i * NDIM, hbuf);
    gemm(hbuf, qkvwb + (long long)i * NDIM * NQKV, nullptr, nullptr, qkv, nullptr,
         MROWS, NQKV, NDIM, NDIM, NQKV, NQKV, 0);
    attn_kernel<<<dim3(BATCH * NHEADS), dim3(256), 0, stream>>>(qkv, obuf);
    gemm(obuf, outwb + (long long)i * NDIM * NDIM, out_b + (long long)i * NDIM, x, x, nullptr,
         MROWS, NDIM, NDIM, NDIM, NDIM, NDIM, 0);
    ln_kernel<<<dim3(MROWS), dim3(256), 0, stream>>>(x, (long long)NDIM,
        ln2_g + (long long)i * NDIM, ln2_b + (long long)i * NDIM, hbuf);
    gemm(hbuf, fc1wb + (long long)i * NDIM * NMLP, fc1_b + (long long)i * NMLP, nullptr,
         nullptr, tbuf, MROWS, NMLP, NDIM, NDIM, NMLP, NMLP, 1);
    gemm(tbuf, fc2wb + (long long)i * NMLP * NDIM, fc2_b + (long long)i * NDIM, x, x, nullptr,
         MROWS, NDIM, NMLP, NMLP, NDIM, NDIM, 0);
  }

  ln_kernel<<<dim3(BATCH), dim3(256), 0, stream>>>(x, (long long)SEQ * NDIM, lnf_g, lnf_b, acls);
  gemm(acls, hdwb, head_b, nullptr, outp, nullptr,
       BATCH, NCLSP, NDIM, NDIM, NCLSR, NCLSR, 0);
}
